// Frames_SDF_VisualAttention_28166395527585
// MI455X (gfx1250) — matrix operand order repaired and checked by reading, not compiled, not run on hardware
//
#include <hip/hip_runtime.h>
#include <hip/hip_bf16.h>

// ---------------- problem constants ----------------
#define NTOK   5120     // N
#define NSDF   4096
#define DMODEL 64       // S
#define NHEAD  8        // HP
#define EDIM   32       // E
#define NBCH   80       // NB
#define OCH    256      // S*FC
#define NPIX   4096     // 64*64
#define QBLK   128      // queries per block (8 waves x 16)
#define NQB    40       // 5120/128
#define NKC    (NTOK / 32)

typedef __attribute__((ext_vector_type(16))) _Float16 v16h;
typedef __attribute__((ext_vector_type(8)))  _Float16 v8h;
typedef __attribute__((ext_vector_type(8)))  float    v8f;

#if defined(__gfx1250__) && __has_builtin(__builtin_amdgcn_global_load_async_to_lds_b128)
#define HAVE_ASYNC_LDS 1
// exact parameter types for the async-to-LDS builtin (from hipcc diagnostics):
// (int4 AS(1)*, int4 AS(3)*, Imm int offset, Imm int cpol)
typedef int async_b128_t __attribute__((vector_size(16)));
typedef __attribute__((address_space(1))) async_b128_t* gptr128;
typedef __attribute__((address_space(3))) async_b128_t* lptr128;
#else
#define HAVE_ASYNC_LDS 0
#endif

// ---------------- workspace layout (bytes) ----------------
static constexpr size_t OFF_X    = 0;                         // 5120*64*4
static constexpr size_t OFF_XH   = OFF_X    + 1310720;        // 5120*64*2
static constexpr size_t OFF_QH   = OFF_XH   + 655360;         // 8*5120*32*2
static constexpr size_t OFF_KH   = OFF_QH   + 2621440;
static constexpr size_t OFF_VT   = OFF_KH   + 2621440;
static constexpr size_t OFF_OH   = OFF_VT   + 2621440;        // 5120*256*2
static constexpr size_t OFF_XN   = OFF_OH   + 2621440;        // 5120*64*4
static constexpr size_t OFF_WQT  = OFF_XN   + 1310720;        // 8*32*64*2
static constexpr size_t OFF_WKT  = OFF_WQT  + 32768;
static constexpr size_t OFF_WVT  = OFF_WKT  + 32768;
static constexpr size_t OFF_WOT  = OFF_WVT  + 32768;          // 64*256*2
static constexpr size_t OFF_STAT = OFF_WOT  + 32768;          // 2 f32
static constexpr size_t OFF_PART = OFF_STAT + 256;            // 320*2 f32

// ---------------- WMMA fragment helpers (16x16x32 f16) ----------------
// A (16x32, MxK), lane l: row m=l&15, k = (i&7) + 8*(l>>4) + 16*(i>>3)
__device__ __forceinline__ v16h load_frag_a(const _Float16* p, int ld, int lane) {
    const int m = lane & 15, hi = lane >> 4;
    const _Float16* r0 = p + m * ld + hi * 8;
    v8h lo = *(const v8h*)(r0);
    v8h hh = *(const v8h*)(r0 + 16);
    return __builtin_shufflevector(lo, hh, 0,1,2,3,4,5,6,7,8,9,10,11,12,13,14,15);
}
// B (32x16, KxN) from "n-major, k-contiguous" storage: lane l: col n=l&15,
// k = (i&7) + 8*(l>>4) + 16*(i>>3)
__device__ __forceinline__ v16h load_frag_b(const _Float16* p, int ld, int lane) {
    const int n = lane & 15, hi = lane >> 4;
    const _Float16* r0 = p + n * ld + hi * 8;
    v8h lo = *(const v8h*)(r0);
    v8h hh = *(const v8h*)(r0 + 16);
    return __builtin_shufflevector(lo, hh, 0,1,2,3,4,5,6,7,8,9,10,11,12,13,14,15);
}
#define WMMA_F16(a, b, c) \
    __builtin_amdgcn_wmma_f32_16x16x32_f16(false, (a), false, (b), (short)0, (c), false, false)

__device__ __forceinline__ float red_max16(float v) {
    #pragma unroll
    for (int m = 1; m < 16; m <<= 1) v = fmaxf(v, __shfl_xor(v, m, 16));
    return v;
}
__device__ __forceinline__ float red_sum16(float v) {
    #pragma unroll
    for (int m = 1; m < 16; m <<= 1) v += __shfl_xor(v, m, 16);
    return v;
}

// ---------------- kernel 1: weight prep (f32 -> f16, transposed) ----------------
__global__ __launch_bounds__(256) void prep_weights(
    const float* __restrict__ WQ, const float* __restrict__ WK,
    const float* __restrict__ WV, const float* __restrict__ Wout,
    _Float16* __restrict__ WQt, _Float16* __restrict__ WKt,
    _Float16* __restrict__ WVt, _Float16* __restrict__ Wot) {
    int i = blockIdx.x * 256 + threadIdx.x;
    if (i < NHEAD * DMODEL * EDIM) {           // (h,d,e) -> [h][e][d]
        int h = i / (DMODEL * EDIM);
        int r = i % (DMODEL * EDIM);
        int d = r / EDIM, e = r % EDIM;
        int o = (h * EDIM + e) * DMODEL + d;
        WQt[o] = (_Float16)WQ[i];
        WKt[o] = (_Float16)WK[i];
        WVt[o] = (_Float16)WV[i];
    }
    if (i < OCH * DMODEL) {                    // (k,n) -> [n][k]
        int k = i / DMODEL, n = i % DMODEL;
        Wot[n * OCH + k] = (_Float16)Wout[i];
    }
}

// ---------------- kernel 2: patch embed + pos -> X ----------------
__global__ __launch_bounds__(64) void build_x(
    const float* __restrict__ cond1, const float* __restrict__ cond2,
    const float* __restrict__ W_emb, const float* __restrict__ b_emb,
    const float* __restrict__ W_c3,  const float* __restrict__ b_c3,
    const float* __restrict__ pos, float* __restrict__ X, _Float16* __restrict__ Xh) {
    const int n = blockIdx.x, t = threadIdx.x;
    __shared__ float patch[64];
    float val;
    if (n < NSDF) {
        int g0 = n >> 8, g1 = (n >> 4) & 15, g2 = n & 15;
        int p0 = t >> 4, p1 = (t >> 2) & 3, p2 = t & 3;
        val = cond1[((g0 * 4 + p0) * 64 + (g1 * 4 + p1)) * 64 + (g2 * 4 + p2)];
    } else {
        int nf = n - NSDF;
        int co = nf >> 8, gg = nf & 255, g0 = gg >> 4, g1 = gg & 15;
        int ci = t >> 4, p0 = (t >> 2) & 3, p1 = t & 3;
        val = cond2[((co * 4 + ci) * 64 + (g0 * 4 + p0)) * 64 + (g1 * 4 + p1)];
    }
    patch[t] = val;
    __syncthreads();
    const float* Wrow = (n < NSDF ? W_c3 : W_emb) + t * 64;
    float acc = (n < NSDF ? b_c3[t] : b_emb[t]);
    #pragma unroll 8
    for (int k = 0; k < 64; ++k) acc += patch[k] * Wrow[k];
    acc += pos[n * 64 + t];
    X[n * 64 + t] = acc;
    Xh[n * 64 + t] = (_Float16)acc;
}

// ---------------- kernel 3: QKV projection (WMMA) ----------------
__device__ __forceinline__ void qkv_tile(v16h a0, v16h a1, const _Float16* wb,
                                         int lane, v8f& c0, v8f& c1) {
    v16h b00 = load_frag_b(wb, DMODEL, lane);
    v16h b01 = load_frag_b(wb + 32, DMODEL, lane);
    v16h b10 = load_frag_b(wb + 16 * DMODEL, DMODEL, lane);
    v16h b11 = load_frag_b(wb + 16 * DMODEL + 32, DMODEL, lane);
    c0 = v8f{}; c1 = v8f{};
    c0 = WMMA_F16(a0, b00, c0); c0 = WMMA_F16(a1, b01, c0);
    c1 = WMMA_F16(a0, b10, c1); c1 = WMMA_F16(a1, b11, c1);
}

__global__ __launch_bounds__(256) void qkv_kernel(
    const _Float16* __restrict__ Xh,
    const _Float16* __restrict__ WQt, const _Float16* __restrict__ WKt,
    const _Float16* __restrict__ WVt,
    _Float16* __restrict__ Qh, _Float16* __restrict__ Kh, _Float16* __restrict__ Vt) {
    const int head = blockIdx.x / NQB, qb = blockIdx.x % NQB;
    const int wave = threadIdx.x >> 5, lane = threadIdx.x & 31;
    const int nn = lane & 15, hi = lane >> 4;
    const int m0 = qb * QBLK + wave * 16;

    v16h a0 = load_frag_a(Xh + (size_t)m0 * DMODEL, DMODEL, lane);
    v16h a1 = load_frag_a(Xh + (size_t)m0 * DMODEL + 32, DMODEL, lane);

    v8f c0, c1;
    // Q
    qkv_tile(a0, a1, WQt + head * EDIM * DMODEL, lane, c0, c1);
    #pragma unroll
    for (int r = 0; r < 8; ++r) {
        int tok = m0 + r + 8 * hi;
        Qh[((size_t)head * NTOK + tok) * EDIM + nn]      = (_Float16)c0[r];
        Qh[((size_t)head * NTOK + tok) * EDIM + 16 + nn] = (_Float16)c1[r];
    }
    // K
    qkv_tile(a0, a1, WKt + head * EDIM * DMODEL, lane, c0, c1);
    #pragma unroll
    for (int r = 0; r < 8; ++r) {
        int tok = m0 + r + 8 * hi;
        Kh[((size_t)head * NTOK + tok) * EDIM + nn]      = (_Float16)c0[r];
        Kh[((size_t)head * NTOK + tok) * EDIM + 16 + nn] = (_Float16)c1[r];
    }
    // V  (stored transposed: [h][e][token])
    qkv_tile(a0, a1, WVt + head * EDIM * DMODEL, lane, c0, c1);
    #pragma unroll
    for (int r = 0; r < 8; ++r) {
        int tok = m0 + r + 8 * hi;
        Vt[((size_t)head * EDIM + nn) * NTOK + tok]        = (_Float16)c0[r];
        Vt[((size_t)head * EDIM + 16 + nn) * NTOK + tok]   = (_Float16)c1[r];
    }
}

// ---------------- kernel 4: flash attention (WMMA, online softmax) ----------------
// K/V chunks (32 keys) staged in LDS; with async-to-LDS support the staging is
// double-buffered via GLOBAL_LOAD_ASYNC_TO_LDS_B128 (ASYNCcnt) so the next
// chunk's fetch overlaps the current chunk's WMMAs, one barrier per iteration.
__global__ __launch_bounds__(256) void attn_kernel(
    const _Float16* __restrict__ Qh, const _Float16* __restrict__ Kh,
    const _Float16* __restrict__ Vt, _Float16* __restrict__ Oh) {
    const int head = blockIdx.x / NQB, qb = blockIdx.x % NQB;
    const int tid = threadIdx.x;
    const int wave = tid >> 5, lane = tid & 31;
    const int nn = lane & 15, hi = lane >> 4;
    const int q0 = qb * QBLK + wave * 16;
    const float SCALE = 0.17677669529663687f;   // 1/sqrt(32)

    __shared__ _Float16 Kbuf[2][32 * 32];
    __shared__ _Float16 Vbuf[2][32 * 32];
    __shared__ _Float16 Pbuf[8 * 16 * 32];
    _Float16* pw = Pbuf + wave * 512;

    const _Float16* Khead = Kh + (size_t)head * NTOK * EDIM;
    const _Float16* Vhead = Vt + (size_t)head * EDIM * NTOK;

    v16h aq = load_frag_a(Qh + ((size_t)head * NTOK + q0) * EDIM, EDIM, lane);

    float m_st[8], l_st[8];
    v8f o0 = v8f{}, o1 = v8f{};
    #pragma unroll
    for (int r = 0; r < 8; ++r) { m_st[r] = -1e30f; l_st[r] = 0.f; }

#if HAVE_ASYNC_LDS
    // wave-uniform slice assignment: waves 0-3 load K (128 lanes x 16B = 2KB),
    // waves 4-7 load V (32 e-rows x 4 lanes x 16B = 2KB)
    auto issue = [&](int b, int key0) {
        if (wave < 4) {
            const int off = tid * 8;  // halfs
            __builtin_amdgcn_global_load_async_to_lds_b128(
                (gptr128)(Khead + (size_t)key0 * EDIM + off),
                (lptr128)(&Kbuf[b][off]), 0, 0);
        } else {
            const int u = tid - 128;
            const int e = u >> 2, j = (u & 3) * 8;
            __builtin_amdgcn_global_load_async_to_lds_b128(
                (gptr128)(Vhead + (size_t)e * NTOK + key0 + j),
                (lptr128)(&Vbuf[b][e * 32 + j]), 0, 0);
        }
    };
    issue(0, 0);
#endif

    for (int kc = 0; kc < NKC; ++kc) {
        const int key0 = kc * 32;
#if HAVE_ASYNC_LDS
        const int cur = kc & 1;
#if __has_builtin(__builtin_amdgcn_s_wait_asynccnt)
        __builtin_amdgcn_s_wait_asynccnt(0);
#else
        asm volatile("s_wait_asynccnt 0" ::: "memory");
#endif
        __syncthreads();                 // chunk kc visible; prev compute done
        if (kc + 1 < NKC) issue(cur ^ 1, key0 + 32);
        const _Float16* kb = Kbuf[cur];
        const _Float16* vb = Vbuf[cur];
#else
        const _Float16* kb = Kbuf[0];
        const _Float16* vb = Vbuf[0];
        ((uint2*)Kbuf[0])[tid] = ((const uint2*)(Khead + (size_t)key0 * EDIM))[tid];
        {
            int e = tid >> 3, j = (tid & 7) * 4;
            *(uint2*)&Vbuf[0][e * 32 + j] =
                *(const uint2*)&Vhead[(size_t)e * NTOK + key0 + j];
        }
        int nxt = (kc + 1 < NKC) ? key0 + 32 : key0;
        __builtin_prefetch(Khead + (size_t)nxt * EDIM + lane * 32, 0, 1);
        __syncthreads();
#endif

        // scores: Q(16x32) x K^T -> two 16x16 f32 tiles
        v16h bk0 = load_frag_b(kb, 32, lane);
        v16h bk1 = load_frag_b(kb + 16 * 32, 32, lane);
        v8f zero = v8f{};
        v8f s0 = WMMA_F16(aq, bk0, zero);
        v8f s1 = WMMA_F16(aq, bk1, zero);

        // online softmax per row (rows r+8*hi live in 16-lane halves)
        #pragma unroll
        for (int r = 0; r < 8; ++r) {
            float a0 = s0[r] * SCALE, a1 = s1[r] * SCALE;
            float mx = red_max16(fmaxf(a0, a1));
            float mn = fmaxf(m_st[r], mx);
            float alpha = __expf(m_st[r] - mn);
            float p0 = __expf(a0 - mn), p1 = __expf(a1 - mn);
            l_st[r] = l_st[r] * alpha + red_sum16(p0 + p1);
            m_st[r] = mn;
            o0[r] *= alpha; o1[r] *= alpha;
            pw[(r + 8 * hi) * 32 + nn]      = (_Float16)p0;
            pw[(r + 8 * hi) * 32 + 16 + nn] = (_Float16)p1;
        }
        // wave-internal cross-lane exchange through LDS: drain DS before reads
        asm volatile("s_wait_dscnt 0" ::: "memory");

        v16h aP  = load_frag_a(pw, 32, lane);
        v16h bv0 = load_frag_b(vb, 32, lane);
        v16h bv1 = load_frag_b(vb + 16 * 32, 32, lane);
        o0 = WMMA_F16(aP, bv0, o0);
        o1 = WMMA_F16(aP, bv1, o1);
#if !HAVE_ASYNC_LDS
        __syncthreads();
#endif
    }

    // epilogue: O[n][h*32+e] = o/l
    #pragma unroll
    for (int r = 0; r < 8; ++r) {
        float inv = 1.0f / l_st[r];
        int tok = q0 + r + 8 * hi;
        Oh[(size_t)tok * (NHEAD * EDIM) + head * EDIM + nn]      = (_Float16)(o0[r] * inv);
        Oh[(size_t)tok * (NHEAD * EDIM) + head * EDIM + 16 + nn] = (_Float16)(o1[r] * inv);
    }
}

// ---------------- kernel 5: output projection + residual (WMMA) ----------------
__global__ __launch_bounds__(256) void outproj_kernel(
    const _Float16* __restrict__ Oh, const _Float16* __restrict__ Wot,
    const float* __restrict__ X, float* __restrict__ Xn) {
    const int wave = threadIdx.x >> 5, lane = threadIdx.x & 31;
    const int nn = lane & 15, hi = lane >> 4;
    const int m0 = blockIdx.x * QBLK + wave * 16;

    v8f c[4];
    #pragma unroll
    for (int t = 0; t < 4; ++t) c[t] = v8f{};

    #pragma unroll 2
    for (int ks = 0; ks < 8; ++ks) {
        v16h a = load_frag_a(Oh + (size_t)m0 * OCH + ks * 32, OCH, lane);
        #pragma unroll
        for (int t = 0; t < 4; ++t) {
            v16h b = load_frag_b(Wot + (size_t)(t * 16) * OCH + ks * 32, OCH, lane);
            c[t] = WMMA_F16(a, b, c[t]);
        }
    }
    #pragma unroll
    for (int t = 0; t < 4; ++t) {
        #pragma unroll
        for (int r = 0; r < 8; ++r) {
            size_t idx = (size_t)(m0 + r + 8 * hi) * DMODEL + t * 16 + nn;
            Xn[idx] = c[t][r] + X[idx];
        }
    }
}

// ---------------- kernel 6a/6b: deterministic mean/var reduction ----------------
__global__ __launch_bounds__(256) void reduce_stats(const float* __restrict__ x,
                                                    float* __restrict__ part) {
    __shared__ float sa[256], sb[256];
    const int t = threadIdx.x;
    float a = 0.f, b = 0.f;
    for (int i = blockIdx.x * 256 + t; i < NTOK * DMODEL; i += gridDim.x * 256) {
        float v = x[i]; a += v; b += v * v;
    }
    sa[t] = a; sb[t] = b; __syncthreads();
    for (int st = 128; st > 0; st >>= 1) {
        if (t < st) { sa[t] += sa[t + st]; sb[t] += sb[t + st]; }
        __syncthreads();
    }
    if (t == 0) { part[2 * blockIdx.x] = sa[0]; part[2 * blockIdx.x + 1] = sb[0]; }
}

__global__ __launch_bounds__(256) void final_stats(const float* __restrict__ part,
                                                   float* __restrict__ stats) {
    __shared__ float sa[256], sb[256];
    const int t = threadIdx.x;
    float a = 0.f, b = 0.f;
    for (int i = t; i < 320; i += 256) { a += part[2 * i]; b += part[2 * i + 1]; }
    sa[t] = a; sb[t] = b; __syncthreads();
    for (int st = 128; st > 0; st >>= 1) {
        if (t < st) { sa[t] += sa[t + st]; sb[t] += sb[t + st]; }
        __syncthreads();
    }
    if (t == 0) { stats[0] = sa[0]; stats[1] = sb[0]; }
}

// ---------------- kernel 7: layernorm + 1x1 conv over channel dim ----------------
__global__ __launch_bounds__(256) void norm_conv(
    const float* __restrict__ Xn, const float* __restrict__ stats,
    const float* __restrict__ gamma, const float* __restrict__ beta,
    const float* __restrict__ W2, const float* __restrict__ b2,
    float* __restrict__ out) {
    const int p = blockIdx.x;        // pixel within 64x64 plane
    const int t = threadIdx.x;       // output channel 0..255
    __shared__ float nrm[NBCH];
    const float inv_n = 1.0f / (float)(NTOK * DMODEL);
    const float mu  = stats[0] * inv_n;
    const float var = stats[1] * inv_n - mu * mu;
    const float inv = rsqrtf(var + 1e-5f);
    if (t < NBCH) {
        int j = t * NPIX + p;        // flat index (c*64+h)*64+w
        nrm[t] = (Xn[j] - mu) * inv * gamma[j] + beta[j];
    }
    __syncthreads();
    float acc = b2[t];
    const float* wr = W2 + t * NBCH;
    #pragma unroll 8
    for (int c = 0; c < NBCH; ++c) acc += wr[c] * nrm[c];
    out[(size_t)t * NPIX + p] = acc;
}

// ---------------- host launcher ----------------
extern "C" void kernel_launch(void* const* d_in, const int* in_sizes, int n_in,
                              void* d_out, int out_size, void* d_ws, size_t ws_size,
                              hipStream_t stream) {
    const float* cond1  = (const float*)d_in[0];
    const float* cond2  = (const float*)d_in[1];
    const float* W_emb  = (const float*)d_in[2];
    const float* b_emb  = (const float*)d_in[3];
    const float* W_c3   = (const float*)d_in[4];
    const float* b_c3   = (const float*)d_in[5];
    const float* pos    = (const float*)d_in[6];
    const float* WQ     = (const float*)d_in[7];
    const float* WK     = (const float*)d_in[8];
    const float* WV     = (const float*)d_in[9];
    const float* Wout   = (const float*)d_in[10];
    const float* gamma  = (const float*)d_in[11];
    const float* beta   = (const float*)d_in[12];
    const float* W2     = (const float*)d_in[13];
    const float* b2     = (const float*)d_in[14];

    char* ws = (char*)d_ws;
    float*    X    = (float*)(ws + OFF_X);
    _Float16* Xh   = (_Float16*)(ws + OFF_XH);
    _Float16* Qh   = (_Float16*)(ws + OFF_QH);
    _Float16* Kh   = (_Float16*)(ws + OFF_KH);
    _Float16* Vt   = (_Float16*)(ws + OFF_VT);
    _Float16* Oh   = (_Float16*)(ws + OFF_OH);
    float*    Xn   = (float*)(ws + OFF_XN);
    _Float16* WQt  = (_Float16*)(ws + OFF_WQT);
    _Float16* WKt  = (_Float16*)(ws + OFF_WKT);
    _Float16* WVt  = (_Float16*)(ws + OFF_WVT);
    _Float16* Wot  = (_Float16*)(ws + OFF_WOT);
    float*    stat = (float*)(ws + OFF_STAT);
    float*    part = (float*)(ws + OFF_PART);
    float*    out  = (float*)d_out;

    prep_weights<<<64, 256, 0, stream>>>(WQ, WK, WV, Wout, WQt, WKt, WVt, Wot);
    build_x<<<NTOK, 64, 0, stream>>>(cond1, cond2, W_emb, b_emb, W_c3, b_c3,
                                     pos, X, Xh);
    qkv_kernel<<<NHEAD * NQB, 256, 0, stream>>>(Xh, WQt, WKt, WVt, Qh, Kh, Vt);
    attn_kernel<<<NHEAD * NQB, 256, 0, stream>>>(Qh, Kh, Vt, Oh);
    outproj_kernel<<<NQB, 256, 0, stream>>>(Oh, Wot, X, Xn);
    reduce_stats<<<320, 256, 0, stream>>>(Xn, part);
    final_stats<<<1, 256, 0, stream>>>(part, stat);
    norm_conv<<<NPIX, 256, 0, stream>>>(Xn, stat, gamma, beta, W2, b2, out);
}
